// TrainableTernaryLinear_90254442758762
// MI455X (gfx1250) — compile-verified, matrix operand order checked by reading, not run on hardware
//
#include <hip/hip_runtime.h>

// ---------------------------------------------------------------------------
// TrainableTernaryLinear for MI455X (gfx1250, wave32, WMMA)
//   y = w_scale * (X @ T^T) + tern(b),  T = ternarize(W, 0.05*max|W|)
//   X: [M=8192, K=4096] f32, W: [N=4096, K=4096] f32, out: [M, N] f32
//
// Strategy: quantize once, multiply many.
//   prep_w: W -> ternary bf16 T (exact: {-1,0,+1})
//   prep_x: X -> bf16 hi + bf16 lo (x ~= hi + lo, ~16-bit mantissa fidelity)
//   GEMM:   two chained v_wmma_f32_16x16x32_bf16 per tile (hi & lo passes),
//           double-buffered LDS, 128x128x32 block tile, 8 wave32s.
//           Staging uses GLOBAL_LOAD_ASYNC_TO_LDS_B128 (ASYNCcnt) when the
//           toolchain exposes the builtin; else global->VGPR->LDS.
// Fallback (small ws): fused on-the-fly conversion GEMM.
// ---------------------------------------------------------------------------

typedef __attribute__((ext_vector_type(16))) __bf16 v16bf;
typedef __attribute__((ext_vector_type(8)))  __bf16 v8bf;
typedef __attribute__((ext_vector_type(4)))  __bf16 v4bf;
typedef __attribute__((ext_vector_type(8)))  float  v8f;
typedef __attribute__((ext_vector_type(4)))  int    v4i;

#define BM 128
#define BN 128
#define BK 32
#define LDS_STRIDE 40  // bf16 elems; multiple of 8 keeps 16B-aligned b128 LDS ops

#if defined(__has_builtin)
#if __has_builtin(__builtin_amdgcn_global_load_async_to_lds_b128)
#define TTL_ASYNC_LDS 1
#endif
#if __has_builtin(__builtin_amdgcn_s_wait_asynccnt)
#define TTL_HAS_WAIT_ASYNC_BUILTIN 1
#endif
#endif

#define TTL_AS1 __attribute__((address_space(1)))
#define TTL_AS3 __attribute__((address_space(3)))

__device__ __forceinline__ void ttl_wait_async0() {
#if defined(TTL_ASYNC_LDS)
#if defined(TTL_HAS_WAIT_ASYNC_BUILTIN)
  __builtin_amdgcn_s_wait_asynccnt(0);
#else
  asm volatile("s_wait_asynccnt 0x0" ::: "memory");
#endif
#endif
}

__global__ void ttl_init_scalars(unsigned* ws) {
  ws[0] = 0u;  // max|W| bits
  ws[1] = 0u;  // max|b| bits
}

__global__ __launch_bounds__(256) void ttl_absmax(const float* __restrict__ p,
                                                  int n, unsigned* out) {
  __shared__ float red[256];
  float m = 0.0f;
  for (int i = blockIdx.x * blockDim.x + threadIdx.x; i < n;
       i += gridDim.x * blockDim.x)
    m = fmaxf(m, fabsf(p[i]));
  red[threadIdx.x] = m;
  __syncthreads();
  for (int s = 128; s > 0; s >>= 1) {
    if ((int)threadIdx.x < s)
      red[threadIdx.x] = fmaxf(red[threadIdx.x], red[threadIdx.x + s]);
    __syncthreads();
  }
  if (threadIdx.x == 0) atomicMax(out, __float_as_uint(red[0]));
}

__device__ __forceinline__ float tern_f(float v, float delta) {
  // matches jnp.where(p > delta, 1, where(p < -delta, -1, 0))
  return v > delta ? 1.0f : (v < -delta ? -1.0f : 0.0f);
}

// ---- preprocessing: X -> (hi, lo) bf16 ------------------------------------
__global__ __launch_bounds__(256) void ttl_prep_x(const float* __restrict__ X,
                                                  __bf16* __restrict__ Xh,
                                                  __bf16* __restrict__ Xl,
                                                  int n4) {
  for (int i = blockIdx.x * blockDim.x + threadIdx.x; i < n4;
       i += gridDim.x * blockDim.x) {
    const float4 v = ((const float4*)X)[i];
    const __bf16 h0 = (__bf16)v.x, h1 = (__bf16)v.y;
    const __bf16 h2 = (__bf16)v.z, h3 = (__bf16)v.w;
    v4bf hv = {h0, h1, h2, h3};
    v4bf lv = {(__bf16)(v.x - (float)h0), (__bf16)(v.y - (float)h1),
               (__bf16)(v.z - (float)h2), (__bf16)(v.w - (float)h3)};
    ((v4bf*)Xh)[i] = hv;
    ((v4bf*)Xl)[i] = lv;
  }
}

// ---- preprocessing: W -> ternary bf16 -------------------------------------
__global__ __launch_bounds__(256) void ttl_prep_w(
    const float* __restrict__ W, __bf16* __restrict__ T,
    const unsigned* __restrict__ maxbuf, int n4) {
  const float delta = 0.05f * __uint_as_float(maxbuf[0]);
  for (int i = blockIdx.x * blockDim.x + threadIdx.x; i < n4;
       i += gridDim.x * blockDim.x) {
    const float4 v = ((const float4*)W)[i];
    v4bf tv = {(__bf16)tern_f(v.x, delta), (__bf16)tern_f(v.y, delta),
               (__bf16)tern_f(v.z, delta), (__bf16)tern_f(v.w, delta)};
    ((v4bf*)T)[i] = tv;
  }
}

// one 16B chunk: global -> LDS (async if available)
__device__ __forceinline__ void ttl_copy16(const __bf16* __restrict__ gsrc,
                                           __bf16* ldst) {
#if defined(TTL_ASYNC_LDS)
  // builtin signature (probed via diagnostics): arg0 = v4i addrspace(1)* src,
  // arg1 = v4i addrspace(3)* dst, then imm offset, imm cpol.
  __builtin_amdgcn_global_load_async_to_lds_b128(
      (TTL_AS1 v4i*)(void*)const_cast<__bf16*>(gsrc),
      (TTL_AS3 v4i*)(void*)ldst, 0, 0);
#else
  v8bf v = *(const v8bf*)gsrc;
  *(v8bf*)ldst = v;
#endif
}

// ---- lean GEMM: bf16 inputs, double-buffered LDS --------------------------
__global__ __launch_bounds__(256) void ttl_gemm_wmma_pre(
    const __bf16* __restrict__ Xh, const __bf16* __restrict__ Xl,
    const __bf16* __restrict__ Tm, const float* __restrict__ w_scale_p,
    const float* __restrict__ bias, const float* __restrict__ b_scale_p,
    const unsigned* __restrict__ maxbuf, float* __restrict__ out, int M, int N,
    int K) {
  __shared__ alignas(16) __bf16 Ah[2][BM * LDS_STRIDE];
  __shared__ alignas(16) __bf16 Al[2][BM * LDS_STRIDE];
  __shared__ alignas(16) __bf16 Bt[2][BN * LDS_STRIDE];

  const int tid = threadIdx.x;
  const int m0 = blockIdx.y * BM;
  const int n0 = blockIdx.x * BN;

  // 8 wave32s: 4 along M (32 rows each) x 2 along N (64 cols each)
  const int waveId = tid >> 5;
  const int lane   = tid & 31;
  const int wm     = waveId & 3;
  const int wn     = waveId >> 2;
  const int r16    = lane & 15;
  const int h      = lane >> 4;

  // staging: each thread moves 8 bf16 (one b128) per matrix per half-tile
  const int rowS = tid >> 2;        // 0..63
  const int ch   = (tid & 3) * 8;   // element offset 0,8,16,24 within BK=32

  auto stage = [&](int kk, int buf) {
    const int k0 = kk * BK;
#pragma unroll
    for (int p = 0; p < 2; ++p) {
      const int r = p * 64 + rowS;
      ttl_copy16(&Xh[(size_t)(m0 + r) * K + k0 + ch],
                 &Ah[buf][r * LDS_STRIDE + ch]);
      ttl_copy16(&Xl[(size_t)(m0 + r) * K + k0 + ch],
                 &Al[buf][r * LDS_STRIDE + ch]);
      ttl_copy16(&Tm[(size_t)(n0 + r) * K + k0 + ch],
                 &Bt[buf][r * LDS_STRIDE + ch]);
    }
  };

  v8f acc[2][4] = {};
  const int nk = K / BK;

  stage(0, 0);
  ttl_wait_async0();
  __syncthreads();

  for (int kk = 0; kk < nk; ++kk) {
    const int cur = kk & 1;
    if (kk + 1 < nk) stage(kk + 1, cur ^ 1);  // overlap with compute below

    // fragment loads (ISA 16-bit A/B layout): lane L -> row L%16,
    // K chunks [8h..8h+7] and [16+8h..16+8h+7]
    v16bf aHi[2], aLo[2], bF[4];
#pragma unroll
    for (int mi = 0; mi < 2; ++mi) {
      const int row = wm * 32 + mi * 16 + r16;
      v8bf x0 = *(const v8bf*)&Ah[cur][row * LDS_STRIDE + 8 * h];
      v8bf x1 = *(const v8bf*)&Ah[cur][row * LDS_STRIDE + 16 + 8 * h];
      aHi[mi] = __builtin_shufflevector(x0, x1, 0, 1, 2, 3, 4, 5, 6, 7, 8, 9,
                                        10, 11, 12, 13, 14, 15);
      v8bf y0 = *(const v8bf*)&Al[cur][row * LDS_STRIDE + 8 * h];
      v8bf y1 = *(const v8bf*)&Al[cur][row * LDS_STRIDE + 16 + 8 * h];
      aLo[mi] = __builtin_shufflevector(y0, y1, 0, 1, 2, 3, 4, 5, 6, 7, 8, 9,
                                        10, 11, 12, 13, 14, 15);
    }
#pragma unroll
    for (int ni = 0; ni < 4; ++ni) {
      const int col = wn * 64 + ni * 16 + r16;
      v8bf b0 = *(const v8bf*)&Bt[cur][col * LDS_STRIDE + 8 * h];
      v8bf b1 = *(const v8bf*)&Bt[cur][col * LDS_STRIDE + 16 + 8 * h];
      bF[ni] = __builtin_shufflevector(b0, b1, 0, 1, 2, 3, 4, 5, 6, 7, 8, 9,
                                       10, 11, 12, 13, 14, 15);
    }

#pragma unroll
    for (int mi = 0; mi < 2; ++mi) {
#pragma unroll
      for (int ni = 0; ni < 4; ++ni) {
        acc[mi][ni] = __builtin_amdgcn_wmma_f32_16x16x32_bf16(
            false, aHi[mi], false, bF[ni], (short)0, acc[mi][ni], false, false);
        acc[mi][ni] = __builtin_amdgcn_wmma_f32_16x16x32_bf16(
            false, aLo[mi], false, bF[ni], (short)0, acc[mi][ni], false, false);
      }
    }
    ttl_wait_async0();  // staged copies for kk+1 complete before publish
    __syncthreads();
  }

  // epilogue: scale by w_scale, add ternarized bias
  const float wsc     = *w_scale_p;
  const float bsc     = *b_scale_p;
  const float delta_b = 0.05f * __uint_as_float(maxbuf[1]);
#pragma unroll
  for (int ni = 0; ni < 4; ++ni) {
    const int col  = n0 + wn * 64 + ni * 16 + r16;
    const float bv = bias[col];
    const float bt = bv > delta_b ? bsc : (bv < -delta_b ? -bsc : 0.0f);
#pragma unroll
    for (int mi = 0; mi < 2; ++mi) {
      const int rowbase = m0 + wm * 32 + mi * 16 + h * 8;
#pragma unroll
      for (int j = 0; j < 8; ++j)
        out[(size_t)(rowbase + j) * N + col] = wsc * acc[mi][ni][j] + bt;
    }
  }
}

// ---- fallback: fused on-the-fly conversion GEMM (if ws too small) ---------
__global__ __launch_bounds__(256) void ttl_gemm_wmma_fused(
    const float* __restrict__ X, const float* __restrict__ Wm,
    const float* __restrict__ w_scale_p, const float* __restrict__ bias,
    const float* __restrict__ b_scale_p, const unsigned* __restrict__ maxbuf,
    float* __restrict__ out, int M, int N, int K) {
  __shared__ alignas(16) __bf16 As_hi[BM * LDS_STRIDE];
  __shared__ alignas(16) __bf16 As_lo[BM * LDS_STRIDE];
  __shared__ alignas(16) __bf16 Bs[BN * LDS_STRIDE];

  const int tid = threadIdx.x;
  const int m0 = blockIdx.y * BM;
  const int n0 = blockIdx.x * BN;
  const float delta_w = 0.05f * __uint_as_float(maxbuf[0]);

  const int waveId = tid >> 5;
  const int lane   = tid & 31;
  const int wm     = waveId & 3;
  const int wn     = waveId >> 2;
  const int r16    = lane & 15;
  const int h      = lane >> 4;

  const int rowA = tid >> 3;
  const int c4   = (tid & 7) * 4;

  v8f acc[2][4] = {};
  const int nk = K / BK;

  for (int kk = 0; kk < nk; ++kk) {
    const int k0 = kk * BK;
    __syncthreads();
#pragma unroll
    for (int p = 0; p < 4; ++p) {
      const int r = p * 32 + rowA;
      const float4 xv = *(const float4*)(X + (size_t)(m0 + r) * K + k0 + c4);
      const __bf16 h0 = (__bf16)xv.x, h1 = (__bf16)xv.y;
      const __bf16 h2 = (__bf16)xv.z, h3 = (__bf16)xv.w;
      v4bf hv = {h0, h1, h2, h3};
      v4bf lv = {(__bf16)(xv.x - (float)h0), (__bf16)(xv.y - (float)h1),
                 (__bf16)(xv.z - (float)h2), (__bf16)(xv.w - (float)h3)};
      *(v4bf*)&As_hi[r * LDS_STRIDE + c4] = hv;
      *(v4bf*)&As_lo[r * LDS_STRIDE + c4] = lv;
      const float4 wv = *(const float4*)(Wm + (size_t)(n0 + r) * K + k0 + c4);
      v4bf tv = {(__bf16)tern_f(wv.x, delta_w), (__bf16)tern_f(wv.y, delta_w),
                 (__bf16)tern_f(wv.z, delta_w), (__bf16)tern_f(wv.w, delta_w)};
      *(v4bf*)&Bs[r * LDS_STRIDE + c4] = tv;
    }
    __syncthreads();

    v16bf aHi[2], aLo[2], bF[4];
#pragma unroll
    for (int mi = 0; mi < 2; ++mi) {
      const int row = wm * 32 + mi * 16 + r16;
      v8bf x0 = *(const v8bf*)&As_hi[row * LDS_STRIDE + 8 * h];
      v8bf x1 = *(const v8bf*)&As_hi[row * LDS_STRIDE + 16 + 8 * h];
      aHi[mi] = __builtin_shufflevector(x0, x1, 0, 1, 2, 3, 4, 5, 6, 7, 8, 9,
                                        10, 11, 12, 13, 14, 15);
      v8bf y0 = *(const v8bf*)&As_lo[row * LDS_STRIDE + 8 * h];
      v8bf y1 = *(const v8bf*)&As_lo[row * LDS_STRIDE + 16 + 8 * h];
      aLo[mi] = __builtin_shufflevector(y0, y1, 0, 1, 2, 3, 4, 5, 6, 7, 8, 9,
                                        10, 11, 12, 13, 14, 15);
    }
#pragma unroll
    for (int ni = 0; ni < 4; ++ni) {
      const int col = wn * 64 + ni * 16 + r16;
      v8bf b0 = *(const v8bf*)&Bs[col * LDS_STRIDE + 8 * h];
      v8bf b1 = *(const v8bf*)&Bs[col * LDS_STRIDE + 16 + 8 * h];
      bF[ni] = __builtin_shufflevector(b0, b1, 0, 1, 2, 3, 4, 5, 6, 7, 8, 9,
                                       10, 11, 12, 13, 14, 15);
    }
#pragma unroll
    for (int mi = 0; mi < 2; ++mi) {
#pragma unroll
      for (int ni = 0; ni < 4; ++ni) {
        acc[mi][ni] = __builtin_amdgcn_wmma_f32_16x16x32_bf16(
            false, aHi[mi], false, bF[ni], (short)0, acc[mi][ni], false, false);
        acc[mi][ni] = __builtin_amdgcn_wmma_f32_16x16x32_bf16(
            false, aLo[mi], false, bF[ni], (short)0, acc[mi][ni], false, false);
      }
    }
  }

  const float wsc     = *w_scale_p;
  const float bsc     = *b_scale_p;
  const float delta_b = 0.05f * __uint_as_float(maxbuf[1]);
#pragma unroll
  for (int ni = 0; ni < 4; ++ni) {
    const int col  = n0 + wn * 64 + ni * 16 + r16;
    const float bv = bias[col];
    const float bt = bv > delta_b ? bsc : (bv < -delta_b ? -bsc : 0.0f);
#pragma unroll
    for (int mi = 0; mi < 2; ++mi) {
      const int rowbase = m0 + wm * 32 + mi * 16 + h * 8;
#pragma unroll
      for (int j = 0; j < 8; ++j)
        out[(size_t)(rowbase + j) * N + col] = wsc * acc[mi][ni][j] + bt;
    }
  }
}

extern "C" void kernel_launch(void* const* d_in, const int* in_sizes, int n_in,
                              void* d_out, int out_size, void* d_ws,
                              size_t ws_size, hipStream_t stream) {
  const float* x       = (const float*)d_in[0];
  const float* W       = (const float*)d_in[1];
  const float* w_scale = (const float*)d_in[2];
  const float* b       = (const float*)d_in[3];
  const float* b_scale = (const float*)d_in[4];
  float* out           = (float*)d_out;

  const int D_OUT = in_sizes[3];          // 4096
  const int D_IN  = in_sizes[1] / D_OUT;  // 4096
  const int M     = in_sizes[0] / D_IN;   // 8192 (B*S)

  unsigned* maxbuf = (unsigned*)d_ws;  // [0]=max|W| bits, [1]=max|b| bits

  ttl_init_scalars<<<1, 1, 0, stream>>>(maxbuf);
  ttl_absmax<<<512, 256, 0, stream>>>(W, in_sizes[1], maxbuf + 0);
  ttl_absmax<<<8, 256, 0, stream>>>(b, in_sizes[3], maxbuf + 1);

  const size_t nX = (size_t)M * D_IN;       // X elements
  const size_t nW = (size_t)D_OUT * D_IN;   // W elements
  const size_t need = 256 + nX * 2 * 2 + nW * 2;

  dim3 grid(D_OUT / BN, M / BM);  // (32, 64)

  if (ws_size >= need) {
    __bf16* Xh = (__bf16*)((char*)d_ws + 256);
    __bf16* Xl = Xh + nX;
    __bf16* Tm = Xl + nX;
    ttl_prep_w<<<2048, 256, 0, stream>>>(W, Tm, maxbuf, (int)(nW / 4));
    ttl_prep_x<<<4096, 256, 0, stream>>>(x, Xh, Xl, (int)(nX / 4));
    ttl_gemm_wmma_pre<<<grid, 256, 0, stream>>>(Xh, Xl, Tm, w_scale, b,
                                                b_scale, maxbuf, out, M, D_OUT,
                                                D_IN);
  } else {
    ttl_gemm_wmma_fused<<<grid, 256, 0, stream>>>(x, W, w_scale, b, b_scale,
                                                  maxbuf, out, M, D_OUT, D_IN);
  }
}
